// MultiHeadAttention_55654186221650
// MI455X (gfx1250) — compile-verified, matrix operand order checked by reading, not run on hardware
//
#include <hip/hip_runtime.h>
#include <hip/hip_bf16.h>
#include <stdint.h>

// ---------------------------------------------------------------------------
// Problem constants (from reference): bs=4, n=4096, d=1024, H=16, DK=64,
// N_PERSIS=512 -> m = 4608.  GEMMs: (16384 x 1024) x (1024 x 1024).
// ---------------------------------------------------------------------------
#define BS        4
#define SEQ       4096
#define DMODEL    1024
#define NHEAD     16
#define DK        64
#define NPERSIS   512
#define MTOT      (SEQ + NPERSIS)          // 4608
#define ROWS      (BS * SEQ)               // 16384
#define SCL       2.82842712474619f        // 64^(1/4) = 2^1.5

typedef __attribute__((ext_vector_type(16))) _Float16 v16h;
typedef __attribute__((ext_vector_type(8)))  _Float16 h8;
typedef __attribute__((ext_vector_type(4)))  _Float16 h4;
typedef __attribute__((ext_vector_type(8)))  float    v8f;

// ---------------------------------------------------------------------------
// CDNA5 async global->LDS copy (16B per lane).  The LDS destination address
// is the low 32 bits of the generic pointer (flat->LDS truncation, ISA 10.2).
// Tracked by ASYNCcnt; drain with s_wait_asynccnt before the barrier.
// ---------------------------------------------------------------------------
__device__ __forceinline__ void async_ld16(const _Float16* g, _Float16* lds) {
    const unsigned ldsOff = (unsigned)(uintptr_t)lds;
    asm volatile("global_load_async_to_lds_b128 %0, %1, off"
                 :: "v"(ldsOff), "v"(g) : "memory");
}
__device__ __forceinline__ void wait_async0() {
    asm volatile("s_wait_asynccnt 0" ::: "memory");
}

// ---------------------------------------------------------------------------
// Kernel 1: f32 -> f16 weight conversion (x @ W.T wants B-tile layout [n][k],
// i.e. W row-major, so this is a pure copy-convert; no transpose needed).
// ---------------------------------------------------------------------------
__global__ void __launch_bounds__(256) cvt_f32_to_f16(const float* __restrict__ src,
                                                      _Float16* __restrict__ dst) {
    int i = (blockIdx.x * 256 + threadIdx.x) * 4;
    float4 f = *(const float4*)(src + i);
    h4 h = { (_Float16)f.x, (_Float16)f.y, (_Float16)f.z, (_Float16)f.w };
    *(h4*)(dst + i) = h;
}

// ---------------------------------------------------------------------------
// Kernel 2: tiled WMMA GEMM  C[M,N] = A[M,K] * B^T  where B is stored [n][k]
// (row-major W).  Block tile 128x128x32, 8 waves (2x4), wave tile 64x32.
// Double-buffered LDS; f16 tiles staged with GLOBAL_LOAD_ASYNC_TO_LDS_B128,
// f32 A tiles converted through VGPRs.  Fragment layouts per ISA 7.12.2:
//   A 16x32 f16 : lane<16 -> M=lane,   K runs {0..7},{16..23}
//                 lane>=16 -> M=lane-16,K runs {8..15},{24..31}
//   B mirrors A with lane = N column (LDS tile indexed [n][k] -> contiguous).
//   C 16x16 f32 : VGPR r, lane<16 -> (M=r, N=lane); lane>=16 -> (M=r+8,N=lane-16)
// ---------------------------------------------------------------------------
template<bool A_F16, bool C_F32>
__global__ void __launch_bounds__(256) gemm_wmma(const void* __restrict__ Ap,
                                                 const _Float16* __restrict__ B,
                                                 void* __restrict__ Cp,
                                                 int M, int N, int K) {
    __shared__ __align__(16) _Float16 As[2][128][40];   // stride 80B (16B aligned)
    __shared__ __align__(16) _Float16 Bs[2][128][40];

    const int tid  = threadIdx.x;
    const int lane = tid & 31;
    const int wid  = tid >> 5;
    const int wy   = wid >> 2;        // 0..1  (M direction, 64 rows each)
    const int wx   = wid & 3;         // 0..3  (N direction, 32 cols each)
    const int lhi  = lane >> 4;       // lane half
    const int r16  = lane & 15;
    const int kb   = lhi * 8;         // low-K run start: 0 or 8

    const int m0 = blockIdx.y * 128;
    const int n0 = blockIdx.x * 128;

    const int lr = tid >> 1;          // 0..127 : row loaded by this thread
    const int lk = (tid & 1) * 16;    // 0 or 16

    v8f acc[4][2];
    const v8f zero8 = {0.f,0.f,0.f,0.f,0.f,0.f,0.f,0.f};
#pragma unroll
    for (int mt = 0; mt < 4; ++mt)
#pragma unroll
        for (int nt = 0; nt < 2; ++nt) acc[mt][nt] = zero8;

    auto stage = [&](int buf, int kk) {
        // ---- B tile: raw f16, async DMA to LDS ----
        {
            const _Float16* p = B + (size_t)(n0 + lr) * K + kk + lk;
            async_ld16(p,     &Bs[buf][lr][lk]);
            async_ld16(p + 8, &Bs[buf][lr][lk + 8]);
        }
        if constexpr (A_F16) {
            const _Float16* A = (const _Float16*)Ap;
            const _Float16* p = A + (size_t)(m0 + lr) * K + kk + lk;
            async_ld16(p,     &As[buf][lr][lk]);
            async_ld16(p + 8, &As[buf][lr][lk + 8]);
        } else {
            const float* A = (const float*)Ap;
            const float* p = A + (size_t)(m0 + lr) * K + kk + lk;
            float4 f0 = *(const float4*)(p + 0);
            float4 f1 = *(const float4*)(p + 4);
            float4 f2 = *(const float4*)(p + 8);
            float4 f3 = *(const float4*)(p + 12);
            __builtin_prefetch((const void*)(p + 32), 0, 3);   // next K tile, WGP scope
            h8 h0 = { (_Float16)f0.x,(_Float16)f0.y,(_Float16)f0.z,(_Float16)f0.w,
                      (_Float16)f1.x,(_Float16)f1.y,(_Float16)f1.z,(_Float16)f1.w };
            h8 h1 = { (_Float16)f2.x,(_Float16)f2.y,(_Float16)f2.z,(_Float16)f2.w,
                      (_Float16)f3.x,(_Float16)f3.y,(_Float16)f3.z,(_Float16)f3.w };
            *(h8*)&As[buf][lr][lk]     = h0;
            *(h8*)&As[buf][lr][lk + 8] = h1;
        }
    };

    stage(0, 0);

    for (int kk = 0; kk < K; kk += 32) {
        const int buf = (kk >> 5) & 1;
        wait_async0();          // drain this wave's async LDS writes
        __syncthreads();        // make all waves' tile writes visible
        if (kk + 32 < K) stage(buf ^ 1, kk + 32);   // overlap next tile w/ math

        v16h afrag[4], bfrag[2];
#pragma unroll
        for (int mt = 0; mt < 4; ++mt) {
            const int row = wy * 64 + mt * 16 + r16;
            h8 lo = *(const h8*)&As[buf][row][kb];
            h8 hi = *(const h8*)&As[buf][row][kb + 16];
#pragma unroll
            for (int i = 0; i < 8; ++i) { afrag[mt][i] = lo[i]; afrag[mt][i + 8] = hi[i]; }
        }
#pragma unroll
        for (int nt = 0; nt < 2; ++nt) {
            const int col = wx * 32 + nt * 16 + r16;
            h8 lo = *(const h8*)&Bs[buf][col][kb];
            h8 hi = *(const h8*)&Bs[buf][col][kb + 16];
#pragma unroll
            for (int i = 0; i < 8; ++i) { bfrag[nt][i] = lo[i]; bfrag[nt][i + 8] = hi[i]; }
        }
#pragma unroll
        for (int mt = 0; mt < 4; ++mt)
#pragma unroll
            for (int nt = 0; nt < 2; ++nt)
                acc[mt][nt] = __builtin_amdgcn_wmma_f32_16x16x32_f16(
                    false, afrag[mt], false, bfrag[nt],
                    (short)0, acc[mt][nt], false, false);
    }

    // ---- store C per ISA C/D layout ----
#pragma unroll
    for (int mt = 0; mt < 4; ++mt) {
#pragma unroll
        for (int nt = 0; nt < 2; ++nt) {
            const int col = n0 + wx * 32 + nt * 16 + r16;
#pragma unroll
            for (int rr = 0; rr < 8; ++rr) {
                const int row = m0 + wy * 64 + mt * 16 + lhi * 8 + rr;
                if constexpr (C_F32)
                    ((float*)Cp)[(size_t)row * N + col] = acc[mt][nt][rr];
                else
                    ((_Float16*)Cp)[(size_t)row * N + col] = (_Float16)acc[mt][nt][rr];
            }
        }
    }
}

// ---------------------------------------------------------------------------
// Kernel 3: per-(b,h,d) column softmax stats over m = SEQ + NPERSIS rows of
// (k * dk^0.25).  One block per (b,h); thread t: d = t&63, part = t>>6.
// ---------------------------------------------------------------------------
__global__ void __launch_bounds__(256) col_stats(const _Float16* __restrict__ kh,
                                                 const float* __restrict__ MK,
                                                 float* __restrict__ colM,
                                                 float* __restrict__ colS) {
    const int bh = blockIdx.x;
    const int b = bh >> 4, h = bh & 15;
    const int t = threadIdx.x;
    const int d = t & 63, part = t >> 6;
    __shared__ float red[4][64];

    float mx = -3.0e38f;
    for (int m = part; m < MTOT; m += 4) {
        float v = (m < SEQ)
            ? (float)kh[((size_t)(b * SEQ + m)) * DMODEL + h * DK + d]
            : MK[((size_t)h * NPERSIS + (m - SEQ)) * DK + d];
        mx = fmaxf(mx, v * SCL);
    }
    red[part][d] = mx;
    __syncthreads();
    const float gmx = fmaxf(fmaxf(red[0][d], red[1][d]), fmaxf(red[2][d], red[3][d]));
    __syncthreads();

    float s = 0.f;
    for (int m = part; m < MTOT; m += 4) {
        float v = (m < SEQ)
            ? (float)kh[((size_t)(b * SEQ + m)) * DMODEL + h * DK + d]
            : MK[((size_t)h * NPERSIS + (m - SEQ)) * DK + d];
        s += __expf(v * SCL - gmx);
    }
    red[part][d] = s;
    __syncthreads();
    if (part == 0) {
        colM[bh * 64 + d] = gmx;
        colS[bh * 64 + d] = red[0][d] + red[1][d] + red[2][d] + red[3][d];
    }
}

// ---------------------------------------------------------------------------
// Kernel 4: kv[b,h,d,e] = sum_m lamb[m,d] * v[m,e]   (64x64 per (b,h)).
// Tiles of 64 rows staged through LDS; lamb computed in place.
// ---------------------------------------------------------------------------
__global__ void __launch_bounds__(256) kv_accum(const _Float16* __restrict__ kh,
                                                const _Float16* __restrict__ vh,
                                                const float* __restrict__ MK,
                                                const float* __restrict__ MV,
                                                const float* __restrict__ colM,
                                                const float* __restrict__ colS,
                                                float* __restrict__ kvout) {
    const int bh = blockIdx.x;
    const int b = bh >> 4, h = bh & 15;
    const int t = threadIdx.x;

    __shared__ float kt[64][65];
    __shared__ float vt[64][65];
    __shared__ float Md[64], Sd[64];
    if (t < 64) { Md[t] = colM[bh * 64 + t]; Sd[t] = 1.0f / colS[bh * 64 + t]; }

    const int dIdx = t >> 2;            // 0..63 (accumulator row)
    const int e0   = (t & 3) * 16;      // 16 e-columns per thread
    const int lrow = t >> 2;            // loader row
    const int lcb  = (t & 3) * 16;      // loader col base

    float acc[16];
#pragma unroll
    for (int j = 0; j < 16; ++j) acc[j] = 0.f;
    __syncthreads();

    for (int c = 0; c < MTOT / 64; ++c) {
        if (c < SEQ / 64) {
            const int m = c * 64 + lrow;
            const _Float16* kp = kh + ((size_t)(b * SEQ + m)) * DMODEL + h * DK + lcb;
            const _Float16* vp = vh + ((size_t)(b * SEQ + m)) * DMODEL + h * DK + lcb;
#pragma unroll
            for (int i = 0; i < 16; ++i) {
                kt[lrow][lcb + i] = (float)kp[i];
                vt[lrow][lcb + i] = (float)vp[i];
            }
        } else {
            const int p = (c - SEQ / 64) * 64 + lrow;
            const float* kp = MK + ((size_t)h * NPERSIS + p) * DK + lcb;
            const float* vp = MV + ((size_t)h * NPERSIS + p) * DK + lcb;
#pragma unroll
            for (int i = 0; i < 16; ++i) {
                kt[lrow][lcb + i] = kp[i];
                vt[lrow][lcb + i] = vp[i];
            }
        }
        __syncthreads();
        // transform kt -> normalized lamb
#pragma unroll
        for (int i = 0; i < 16; ++i) {
            const int idx = t * 16 + i;
            const int mm = idx >> 6, dc = idx & 63;
            kt[mm][dc] = __expf(kt[mm][dc] * SCL - Md[dc]) * Sd[dc];
        }
        __syncthreads();
        for (int mm = 0; mm < 64; ++mm) {
            const float l = kt[mm][dIdx];
#pragma unroll
            for (int j = 0; j < 16; ++j) acc[j] += l * vt[mm][e0 + j];
        }
        __syncthreads();
    }
#pragma unroll
    for (int j = 0; j < 16; ++j)
        kvout[(size_t)bh * 4096 + dIdx * 64 + e0 + j] = acc[j];
}

// ---------------------------------------------------------------------------
// Kernel 5: beta = softmax(q, feature axis), z = beta @ kv, write y f16 in
// (b, n, h*64+e) layout.  One thread per sequence row.
// ---------------------------------------------------------------------------
__global__ void __launch_bounds__(256) beta_z(const _Float16* __restrict__ qh,
                                              const float* __restrict__ kvin,
                                              _Float16* __restrict__ yh) {
    const int bh = blockIdx.x;
    const int b = bh >> 4, h = bh & 15;
    const int t = threadIdx.x;

    __shared__ float kvs[64][65];
#pragma unroll
    for (int i = 0; i < 16; ++i) {
        const int idx = t * 16 + i;
        kvs[idx >> 6][idx & 63] = kvin[(size_t)bh * 4096 + idx];
    }
    __syncthreads();

    const int row = blockIdx.y * 256 + t;                 // 0..4095
    const _Float16* qp = qh + ((size_t)(b * SEQ + row)) * DMODEL + h * DK;

    float qv[64];
#pragma unroll
    for (int d = 0; d < 64; ++d) qv[d] = (float)qp[d];
    float mx = -3.0e38f;
#pragma unroll
    for (int d = 0; d < 64; ++d) mx = fmaxf(mx, qv[d]);
    float s = 0.f;
#pragma unroll
    for (int d = 0; d < 64; ++d) { qv[d] = __expf(qv[d] - mx); s += qv[d]; }
    const float inv = 1.0f / s;

    _Float16* yp = yh + ((size_t)(b * SEQ + row)) * DMODEL + h * DK;
    for (int e = 0; e < 64; ++e) {
        float z = 0.f;
#pragma unroll
        for (int d = 0; d < 64; ++d) z += qv[d] * kvs[d][e];
        yp[e] = (_Float16)(z * inv);
    }
}

// ---------------------------------------------------------------------------
// Host launcher.  Workspace layout (bytes):
//   0   : WhQ (2MB) | 2MB: WhK | 4MB: WhV | 6MB: WhO
//   8MB : qh (32MB) | 40MB: kh | 72MB: vh | 104MB: yh
//   136MB: colM (16KB) | +16KB: colS (16KB) | +32KB: kv (1MB)
// ---------------------------------------------------------------------------
extern "C" void kernel_launch(void* const* d_in, const int* in_sizes, int n_in,
                              void* d_out, int out_size, void* d_ws, size_t ws_size,
                              hipStream_t stream) {
    const float* Q  = (const float*)d_in[0];
    const float* K  = (const float*)d_in[1];
    const float* V  = (const float*)d_in[2];
    const float* WQ = (const float*)d_in[3];
    const float* WK = (const float*)d_in[4];
    const float* WV = (const float*)d_in[5];
    const float* WO = (const float*)d_in[6];
    const float* MK = (const float*)d_in[7];
    const float* MV = (const float*)d_in[8];
    float* out = (float*)d_out;

    const size_t MB = 1ull << 20;
    uint8_t* ws = (uint8_t*)d_ws;
    _Float16* WhQ = (_Float16*)(ws + 0 * MB);
    _Float16* WhK = (_Float16*)(ws + 2 * MB);
    _Float16* WhV = (_Float16*)(ws + 4 * MB);
    _Float16* WhO = (_Float16*)(ws + 6 * MB);
    _Float16* qh  = (_Float16*)(ws + 8 * MB);
    _Float16* kh  = (_Float16*)(ws + 40 * MB);
    _Float16* vh  = (_Float16*)(ws + 72 * MB);
    _Float16* yh  = (_Float16*)(ws + 104 * MB);
    float* colM = (float*)(ws + 136 * MB);
    float* colS = (float*)(ws + 136 * MB + (16 << 10));
    float* kvb  = (float*)(ws + 136 * MB + (32 << 10));

    const dim3 blk(256);
    const dim3 gCvt(DMODEL * DMODEL / (256 * 4));          // 1024 blocks
    hipLaunchKernelGGL(cvt_f32_to_f16, gCvt, blk, 0, stream, WQ, WhQ);
    hipLaunchKernelGGL(cvt_f32_to_f16, gCvt, blk, 0, stream, WK, WhK);
    hipLaunchKernelGGL(cvt_f32_to_f16, gCvt, blk, 0, stream, WV, WhV);
    hipLaunchKernelGGL(cvt_f32_to_f16, gCvt, blk, 0, stream, WO, WhO);

    const dim3 gGemm(DMODEL / 128, ROWS / 128);            // (8, 128)
    hipLaunchKernelGGL((gemm_wmma<false, false>), gGemm, blk, 0, stream,
                       (const void*)Q, WhQ, (void*)qh, ROWS, DMODEL, DMODEL);
    hipLaunchKernelGGL((gemm_wmma<false, false>), gGemm, blk, 0, stream,
                       (const void*)K, WhK, (void*)kh, ROWS, DMODEL, DMODEL);
    hipLaunchKernelGGL((gemm_wmma<false, false>), gGemm, blk, 0, stream,
                       (const void*)V, WhV, (void*)vh, ROWS, DMODEL, DMODEL);

    hipLaunchKernelGGL(col_stats, dim3(BS * NHEAD), blk, 0, stream, kh, MK, colM, colS);
    hipLaunchKernelGGL(kv_accum, dim3(BS * NHEAD), blk, 0, stream,
                       kh, vh, MK, MV, colM, colS, kvb);
    hipLaunchKernelGGL(beta_z, dim3(BS * NHEAD, SEQ / 256), blk, 0, stream, qh, kvb, yh);

    hipLaunchKernelGGL((gemm_wmma<true, true>), gGemm, blk, 0, stream,
                       (const void*)yh, WhO, (void*)out, ROWS, DMODEL, DMODEL);
}
